// MSBlock_78340203479507
// MI455X (gfx1250) — matrix operand order repaired and checked by reading, not compiled, not run on hardware
//
#include <hip/hip_runtime.h>
#include <hip/hip_bf16.h>

typedef __attribute__((ext_vector_type(16))) _Float16 v16h;
typedef __attribute__((ext_vector_type(2)))  _Float16 v2h;
typedef __attribute__((ext_vector_type(8)))  float    v8f;

#define CCH 128
#define HH  128
#define WWD 128
#define BB  4
#define NTAP 9
#define HW  (HH * WWD)
#define KCONV (CCH * NTAP)            // 1152
#define WBUF_PER_CONV (KCONV * CCH)   // 147456 f16 per conv

static __device__ __forceinline__ int iclamp(int v, int lo, int hi) {
    return v < lo ? lo : (v > hi ? hi : v);
}

// ---------------------------------------------------------------------------
// Kernel 1: gating — per-batch channel means, 2-wide FC, softmax.
// ---------------------------------------------------------------------------
__global__ void ms_gate(const float* __restrict__ x,
                        const float* __restrict__ fc_w,
                        const float* __restrict__ fc_b,
                        float* __restrict__ gate) {
    __shared__ float pool[CCH];
    const int b = blockIdx.x;
    const int c = threadIdx.x;                    // 128 threads
    const float4* p4 = (const float4*)(x + ((size_t)b * CCH + c) * HW);
    float s = 0.f;
    for (int i = 0; i < HW / 4; ++i) {
        float4 v = p4[i];
        s += v.x + v.y + v.z + v.w;
    }
    pool[c] = s * (1.0f / HW);
    __syncthreads();
    if (c == 0) {
        float z0 = fc_b[0], z1 = fc_b[1];
        for (int i = 0; i < CCH; ++i) {
            z0 += pool[i] * fc_w[i];
            z1 += pool[i] * fc_w[CCH + i];
        }
        float m  = fmaxf(z0, z1);
        float e0 = expf(z0 - m), e1 = expf(z1 - m);
        float inv = 1.0f / (e0 + e1);
        gate[b * 2 + 0] = e0 * inv;
        gate[b * 2 + 1] = e1 * inv;
    }
}

// ---------------------------------------------------------------------------
// Kernel 2: 18-channel offset conv, plain fp32 (weights uniform per wave ->
// scalar loads; x reads coalesced across the 128-thread row).
// ---------------------------------------------------------------------------
__global__ void ms_offconv(const float* __restrict__ x,
                           const float* __restrict__ off_w,
                           const float* __restrict__ off_b,
                           float* __restrict__ off) {
    const int bh = blockIdx.x;                    // B*H blocks
    const int b = bh / HH, h = bh % HH;
    const int w = threadIdx.x;                    // 128 threads = one row
    float acc[18];
#pragma unroll
    for (int oc = 0; oc < 18; ++oc) acc[oc] = off_b[oc];

    for (int t = 0; t < NTAP; ++t) {
        const int ti = t / 3, tj = t % 3;
        const int y  = h + ti - 1;
        if (y < 0 || y >= HH) continue;           // uniform per block
        const int xx = w + tj - 1;
        const float okx = (xx >= 0 && xx < WWD) ? 1.f : 0.f;
        const int xc = iclamp(xx, 0, WWD - 1);
        const float* xrow = x + (((size_t)b * CCH) * HH + y) * WWD;
        for (int ci = 0; ci < CCH; ++ci) {
            float xv = okx * xrow[(size_t)ci * HW + xc];
#pragma unroll
            for (int oc = 0; oc < 18; ++oc)
                acc[oc] = fmaf(off_w[((oc * CCH + ci) * 3 + ti) * 3 + tj], xv, acc[oc]);
        }
    }
#pragma unroll
    for (int oc = 0; oc < 18; ++oc)
        off[(((size_t)b * 18 + oc) * HH + h) * WWD + w] = acc[oc];
}

// ---------------------------------------------------------------------------
// Kernel 3: pre-convert conv_w / def_w to f16 in WMMA B-lane layout:
//   wbuf[ conv ][ t ][ cb ][ wchunk ][ lane ][ slot ]
// lane = hb*16+n holds column N=n, K = hb*16 + slot of the 32-wide K block.
// ---------------------------------------------------------------------------
__global__ void ms_wprep(const float* __restrict__ conv_w,
                         const float* __restrict__ def_w,
                         _Float16* __restrict__ wbuf) {
    const int idx = blockIdx.x * blockDim.x + threadIdx.x;   // < 294912
    const int conv = idx / WBUF_PER_CONV;
    const int r    = idx % WBUF_PER_CONV;
    const int t    = r / 16384;
    const int r2   = r % 16384;
    const int cb   = r2 / 4096;
    const int r3   = r2 % 4096;
    const int wch  = r3 / 512;
    const int e    = r3 % 512;
    const int lane = e / 16, slot = e % 16;
    const int n  = lane & 15, hb = lane >> 4;
    const int o  = wch * 16 + n;
    const int ci = cb * 32 + hb * 16 + slot;
    const float* src = conv ? def_w : conv_w;
    wbuf[idx] = (_Float16)src[((o * CCH + ci) * 3 + t / 3) * 3 + t % 3];
}

// ---------------------------------------------------------------------------
// Kernel 4: fused lx (dense conv) + gx (deform conv) + gate-mix + ReLU.
// One workgroup = 16 pixels x 128 out-channels; 8 waves, each wave owns a
// 16x16 tile of both outputs; K = 1152 walked in 36 steps of 32 via
// v_wmma_f32_16x16x32_f16. Thread tid handles pixel m = tid&15 and the
// K-pair k = 2*(tid>>4)+{0,1} -> coalesced x reads + one packed b32 LDS
// store per conv, one shared bilinear table entry per tap. B fragments are
// software-pipelined one step ahead in registers so the pre-WMMA wait is
// only the short LDS A-tile dependency.
// ---------------------------------------------------------------------------
__global__ void __launch_bounds__(256)
ms_fused(const float* __restrict__ x,
         const _Float16* __restrict__ wbuf,
         const float* __restrict__ off,
         const float* __restrict__ gate,
         const float* __restrict__ conv_b,
         const float* __restrict__ def_b,
         float* __restrict__ out) {
    __shared__ __align__(16) float tabW[144][4];   // masked bilinear weights
    __shared__ __align__(16) int   tabC[144][4];   // clamped y0,y1,x0,x1
    __shared__ __align__(32) _Float16 Al[2][512];  // dense-conv A tile (lane-packed)
    __shared__ __align__(32) _Float16 Ag[2][512];  // deform A tile (lane-packed)
    __shared__ float so[8 * 256];                  // store staging

    const int bid = blockIdx.x;                    // 4 * 128 * 8 = 4096
    const int b   = bid >> 10;
    const int h   = (bid >> 3) & 127;
    const int w0  = (bid & 7) << 4;
    const int tid = threadIdx.x;
    const int wave = tid >> 5, lane = tid & 31;

    // ---- per-(pixel,tap) bilinear coefficient table -----------------------
    if (tid < 144) {
        const int t = tid >> 4, m = tid & 15;
        const float dy = off[(((size_t)b * 18 + 2 * t    ) * HH + h) * WWD + (w0 + m)];
        const float dx = off[(((size_t)b * 18 + 2 * t + 1) * HH + h) * WWD + (w0 + m)];
        const float py = (float)(h + t / 3 - 1) + dy;
        const float px = (float)(w0 + m + t % 3 - 1) + dx;
        const float fy = floorf(py), fx = floorf(px);
        const int y0 = (int)fy, x0 = (int)fx;
        const float wy1 = py - fy, wx1 = px - fx;
        const float wy0 = 1.f - wy1, wx0 = 1.f - wx1;
        const float my0 = (y0     >= 0 && y0     < HH)  ? 1.f : 0.f;
        const float my1 = (y0 + 1 >= 0 && y0 + 1 < HH)  ? 1.f : 0.f;
        const float mx0 = (x0     >= 0 && x0     < WWD) ? 1.f : 0.f;
        const float mx1 = (x0 + 1 >= 0 && x0 + 1 < WWD) ? 1.f : 0.f;
        tabW[tid][0] = wy0 * wx0 * my0 * mx0;
        tabW[tid][1] = wy0 * wx1 * my0 * mx1;
        tabW[tid][2] = wy1 * wx0 * my1 * mx0;
        tabW[tid][3] = wy1 * wx1 * my1 * mx1;
        tabC[tid][0] = iclamp(y0,     0, HH  - 1);
        tabC[tid][1] = iclamp(y0 + 1, 0, HH  - 1);
        tabC[tid][2] = iclamp(x0,     0, WWD - 1);
        tabC[tid][3] = iclamp(x0 + 1, 0, WWD - 1);
    }
    __syncthreads();

    // ---- per-thread fixed element mapping ---------------------------------
    // pixel m, K-pair base k0 = 2q (q = tid>>4); both elements share (m, tap)
    const int m  = tid & 15;
    const int q  = tid >> 4;
    const int k0 = 2 * q;
    const int hb = (k0 >> 4) & 1;                  // bit4 of k0 == bit4 of k0+1
    const int jj = k0 & 15;                        // even slot
    const int e0 = (m + 16 * hb) * 16 + jj;        // packed b32 target (4B aligned)

    const float* xb = x + (size_t)b * CCH * HW;    // batch base
    _Float16* al_st[2] = { &Al[0][e0], &Al[1][e0] };
    _Float16* ag_st[2] = { &Ag[0][e0], &Ag[1][e0] };
    const _Float16* al_ld[2] = { &Al[0][lane * 16], &Al[1][lane * 16] };
    const _Float16* ag_ld[2] = { &Ag[0][lane * 16], &Ag[1][lane * 16] };

    // ---- B fragment pipeline: current step held in registers --------------
    const _Float16* blp = wbuf + (size_t)wave * 512 + (size_t)lane * 16;
    const _Float16* bgp = blp + WBUF_PER_CONV;
    v16h bl_cur = *(const v16h*)blp;
    v16h bg_cur = *(const v16h*)bgp;

    v8f accl = {};
    v8f accg = {};

    for (int t = 0; t < NTAP; ++t) {
        // ---- hoisted per-tap geometry (shared by both K elements) ---------
        const int tm = t * 16 + m;
        const int4   cc = *(const int4*)  &tabC[tm][0];
        const float4 ww = *(const float4*)&tabW[tm][0];
        const int g00 = cc.x * WWD + cc.z;
        const int g01 = cc.x * WWD + cc.w;
        const int g10 = cc.y * WWD + cc.z;
        const int g11 = cc.y * WWD + cc.w;

        const int y  = h + t / 3 - 1;
        const int xx = w0 + m + t % 3 - 1;
        const float dmask = (y >= 0 && y < HH && xx >= 0 && xx < WWD) ? 1.f : 0.f;
        const int   dofs  = iclamp(y, 0, HH - 1) * WWD + iclamp(xx, 0, WWD - 1);

#pragma unroll
        for (int cb = 0; cb < 4; ++cb) {
            const int buf = cb & 1;                // == (t*4+cb)&1

            // issue next step's B loads now so they overlap build + barrier
            // (workspace has one step of slack past the real weight data)
            blp += 4096;
            bgp += 4096;
            const v16h bl_nxt = *(const v16h*)blp;
            const v16h bg_nxt = *(const v16h*)bgp;
            __builtin_prefetch(blp + 4096, 0, 3);  // distance-2 line prefetch
            __builtin_prefetch(bgp + 4096, 0, 3);

            const float* c0 = xb + (size_t)(cb * 32 + k0) * HW;
            const float* c1 = c0 + HW;

            // dense conv samples (branchless zero-pad)
            v2h pl;
            pl.x = (_Float16)(dmask * c0[dofs]);
            pl.y = (_Float16)(dmask * c1[dofs]);
            *(v2h*)al_st[buf] = pl;

            // deformable bilinear samples
            v2h pg;
            pg.x = (_Float16)(ww.x * c0[g00] + ww.y * c0[g01] +
                              ww.z * c0[g10] + ww.w * c0[g11]);
            pg.y = (_Float16)(ww.x * c1[g00] + ww.y * c1[g01] +
                              ww.z * c1[g10] + ww.w * c1[g11]);
            *(v2h*)ag_st[buf] = pg;

            __syncthreads();

            const v16h a_l = *(const v16h*)al_ld[buf];
            const v16h a_g = *(const v16h*)ag_ld[buf];

            accl = __builtin_amdgcn_wmma_f32_16x16x32_f16(false, a_l, false, bl_cur,
                                                          (short)0, accl, false, false);
            accg = __builtin_amdgcn_wmma_f32_16x16x32_f16(false, a_g, false, bg_cur,
                                                          (short)0, accg, false, false);
            bl_cur = bl_nxt;
            bg_cur = bg_nxt;
        }
    }

    // ---- epilogue: bias + gate mix + ReLU, staged for coalesced stores ----
    const float wl = gate[b * 2 + 0];
    const float wg = gate[b * 2 + 1];
    const int n  = lane & 15;
    const int o  = wave * 16 + n;
    const float cbv = conv_b[o], dbv = def_b[o];
    const int mh = (lane >> 4) * 8;
#pragma unroll
    for (int r = 0; r < 8; ++r) {
        const int mm = r + mh;                     // C/D layout: VGPR r, M = r + 8*(lane/16)
        const float v = wl * (accl[r] + cbv) + wg * (accg[r] + dbv);
        so[wave * 256 + n * 16 + mm] = fmaxf(v, 0.f);
    }
    __syncthreads();
#pragma unroll
    for (int i = 0; i < 8; ++i) {
        const int idx = i * 32 + lane;
        const int nn = idx >> 4, mm = idx & 15;
        out[(((size_t)b * CCH + wave * 16 + nn) * HH + h) * WWD + (w0 + mm)] =
            so[wave * 256 + idx];
    }
}

// ---------------------------------------------------------------------------
extern "C" void kernel_launch(void* const* d_in, const int* in_sizes, int n_in,
                              void* d_out, int out_size, void* d_ws, size_t ws_size,
                              hipStream_t stream) {
    const float* x      = (const float*)d_in[0];
    const float* conv_w = (const float*)d_in[1];
    const float* conv_b = (const float*)d_in[2];
    const float* off_w  = (const float*)d_in[3];
    const float* off_b  = (const float*)d_in[4];
    const float* def_w  = (const float*)d_in[5];
    const float* def_b  = (const float*)d_in[6];
    const float* fc_w   = (const float*)d_in[7];
    const float* fc_b   = (const float*)d_in[8];
    float* out = (float*)d_out;

    char* ws = (char*)d_ws;
    float*    gate  = (float*)ws;                          // 8 floats
    float*    offws = (float*)(ws + 256);                  // 4*18*128*128 f32 = 4,718,592 B
    _Float16* wbuf  = (_Float16*)(ws + 256 + 4718592);     // 2*147456 f16 + 8KB pipeline slack

    ms_gate   <<<BB,      CCH, 0, stream>>>(x, fc_w, fc_b, gate);
    ms_offconv<<<BB * HH, WWD, 0, stream>>>(x, off_w, off_b, offws);
    ms_wprep  <<<1152,    256, 0, stream>>>(conv_w, def_w, wbuf);
    ms_fused  <<<4096,    256, 0, stream>>>(x, wbuf, offws, gate, conv_b, def_b, out);
}
